// EBlock_45853070852214
// MI455X (gfx1250) — matrix operand order repaired and checked by reading, not compiled, not run on hardware
//
#include <hip/hip_runtime.h>
#include <math.h>

// ---------------------------------------------------------------------------
// Graph E-block for MI455X (gfx1250, wave32, WMMA).
//   hv  = LN(GELU(node @ W_node))            [50000,128]
//   he  = exp(LN(edge @ W_edge))             [800000,128]  (never materialized)
//   h   = segment_sum(hv[src]*he, dst)       [50000,128]   (L2-resident atomics)
//   out = LN(GELU(h @ W_out))                [50000,16]
// GEMMs run on v_wmma_f32_16x16x32_bf16 (fp32 accum, bf16 operands).
// Weights are staged in LDS pre-swizzled so a B fragment is ONE contiguous
// 32-byte load per lane (2x ds_load_b128) instead of 16x ds_load_u16.
// ---------------------------------------------------------------------------

#define N_NODES 50000
#define N_EDGES 800000
#define NODE_IN 256
#define EDGE_IN 64
#define HID     128
#define OUT_DIM 16
#define LN_EPS  1e-5f

typedef __attribute__((ext_vector_type(16))) __bf16 v16bf;
typedef __attribute__((ext_vector_type(8)))  float  v8f;

__device__ __forceinline__ v8f wmma_bf16(v16bf a, v16bf b, v8f c) {
  // 8 args: (neg_a, A, neg_b, B, c_mod, C, reuse_a, reuse_b)
  return __builtin_amdgcn_wmma_f32_16x16x32_bf16(false, a, false, b,
                                                 (short)0, c, false, false);
}

__device__ __forceinline__ float gelu_exact(float x) {
  return 0.5f * x * (1.0f + erff(x * 0.70710678118654752f));
}

// A-fragment (16x32 bf16) from fp32 row-major memory.
// Contiguous K order used here (same as B packing): lane&15 = row M; half = lane>>4 selects K sub-block.
//   elem i   (i<8): K = kOff + half*16 + i
//   elem 8+i (i<8): K = kOff + half*16 + 8 + i
__device__ __forceinline__ v16bf load_a_frag_f32(const float* __restrict__ row,
                                                 int kOff, int half) {
  const float4* p0 = (const float4*)(row + kOff + half * 16);
  const float4* p1 = (const float4*)(row + kOff + half * 16 + 8);
  float4 a0 = p0[0], a1 = p0[1];
  float4 b0 = p1[0], b1 = p1[1];
  v16bf a;
  a[0]  = (__bf16)a0.x; a[1]  = (__bf16)a0.y; a[2]  = (__bf16)a0.z; a[3]  = (__bf16)a0.w;
  a[4]  = (__bf16)a1.x; a[5]  = (__bf16)a1.y; a[6]  = (__bf16)a1.z; a[7]  = (__bf16)a1.w;
  a[8]  = (__bf16)b0.x; a[9]  = (__bf16)b0.y; a[10] = (__bf16)b0.z; a[11] = (__bf16)b0.w;
  a[12] = (__bf16)b1.x; a[13] = (__bf16)b1.y; a[14] = (__bf16)b1.z; a[15] = (__bf16)b1.w;
  return a;
}

// Stage fp32 weights W[K][NT*16] into LDS, pre-swizzled to B-fragment layout:
//   sW[((kk*NT + t)*32 + lane)*16 + i] = bf16(W[(kk*32 + (lane>>4)*16 + i)][t*16 + (lane&15)])
// so each lane's 16 fragment elements (K = kBase..kBase+15, fixed column) are
// contiguous: the fragment load is one 32-byte LDS vector load.
// K order used here (same as A loader): lane&15 = column; half-wave splits K lo/hi 16;
// VGPR j holds K = kBase+2j (lo16), kBase+2j+1 (hi16)  =>  element i <-> K+i.
template <int K, int NT>
__device__ __forceinline__ void stage_w_packed(const float* __restrict__ W,
                                               __bf16* __restrict__ sW) {
  const int total = K * NT * 16;
  for (int i = threadIdx.x; i < total; i += blockDim.x) {
    const int k    = i / (NT * 16);        // power-of-two divide
    const int n    = i - k * (NT * 16);
    const int t    = n >> 4;
    const int l15  = n & 15;
    const int kk   = k >> 5;
    const int half = (k >> 4) & 1;
    const int j    = k & 15;
    const int lane = (half << 4) | l15;
    sW[(((kk * NT + t) << 5) | lane) * 16 + j] = (__bf16)W[i];
  }
}

__device__ __forceinline__ v16bf load_b_frag_packed(const __bf16* __restrict__ sW,
                                                    int frag, int lane) {
  return *(const v16bf*)(sW + (((frag << 5) | lane) << 4));
}

// ---------------------------------------------------------------------------
// Kernel 1: hv = LN(GELU(node_feats @ W_node))
// One wave = 16 rows x 128 cols (8 N-tiles). 8 waves per block.
// ---------------------------------------------------------------------------
__global__ void __launch_bounds__(256)
node_proj_kernel(const float* __restrict__ X, const float* __restrict__ W,
                 const float* __restrict__ g, const float* __restrict__ b,
                 float* __restrict__ hv) {
  __shared__ alignas(32) __bf16 sW[NODE_IN * HID];  // 64 KB of 320 KB LDS
  stage_w_packed<NODE_IN, 8>(W, sW);
  __syncthreads();

  const int wave  = threadIdx.x >> 5;
  const int lane  = threadIdx.x & 31;
  const int tile  = blockIdx.x * 8 + wave;
  if (tile >= N_NODES / 16) return;  // wave-uniform: EXEC stays all-ones
  const int R     = tile * 16;
  const int col15 = lane & 15;
  const int half  = lane >> 4;

  const float* arow = X + (size_t)(R + col15) * NODE_IN;

  v8f acc[8] = {};
#pragma unroll
  for (int kk = 0; kk < NODE_IN / 32; ++kk) {
    v16bf af = load_a_frag_f32(arow, kk * 32, half);
#pragma unroll
    for (int t = 0; t < 8; ++t) {
      v16bf bf = load_b_frag_packed(sW, kk * 8 + t, lane);
      acc[t] = wmma_bf16(af, bf, acc[t]);
    }
  }

  // GELU then LayerNorm over HID=128. Row of acc-index m is (m + half*8);
  // its 128 values live in acc[t][m] across the 16 lanes of this half-group.
  float s1[8], s2[8];
#pragma unroll
  for (int m = 0; m < 8; ++m) { s1[m] = 0.f; s2[m] = 0.f; }
#pragma unroll
  for (int t = 0; t < 8; ++t)
#pragma unroll
    for (int m = 0; m < 8; ++m) {
      float x = gelu_exact(acc[t][m]);
      acc[t][m] = x;
      s1[m] += x;
      s2[m] += x * x;
    }
#pragma unroll
  for (int m = 0; m < 8; ++m)
#pragma unroll
    for (int mask = 1; mask < 16; mask <<= 1) {  // butterfly within 16-lane half
      s1[m] += __shfl_xor(s1[m], mask, 32);
      s2[m] += __shfl_xor(s2[m], mask, 32);
    }

  float mean[8], rstd[8];
#pragma unroll
  for (int m = 0; m < 8; ++m) {
    mean[m] = s1[m] * (1.0f / HID);
    float var = s2[m] * (1.0f / HID) - mean[m] * mean[m];
    rstd[m] = rsqrtf(var + LN_EPS);
  }
#pragma unroll
  for (int t = 0; t < 8; ++t) {
    const float gg = g[t * 16 + col15];
    const float bb = b[t * 16 + col15];
#pragma unroll
    for (int m = 0; m < 8; ++m) {
      const int r = R + m + half * 8;
      hv[(size_t)r * HID + t * 16 + col15] =
          (acc[t][m] - mean[m]) * rstd[m] * gg + bb;
    }
  }
}

// ---------------------------------------------------------------------------
// Kernel 2: he = exp(LN(edge_feats @ W_edge)); h[dst] += hv[src] * he  (fused)
// ---------------------------------------------------------------------------
__global__ void __launch_bounds__(256)
edge_msg_kernel(const float* __restrict__ Xe, const int* __restrict__ src,
                const int* __restrict__ dst, const float* __restrict__ W,
                const float* __restrict__ g, const float* __restrict__ b,
                const float* __restrict__ hv, float* __restrict__ h) {
  __shared__ alignas(32) __bf16 sW[EDGE_IN * HID];  // 16 KB
  stage_w_packed<EDGE_IN, 8>(W, sW);
  __syncthreads();

  const int wave  = threadIdx.x >> 5;
  const int lane  = threadIdx.x & 31;
  const int tile  = blockIdx.x * 8 + wave;
  if (tile >= N_EDGES / 16) return;
  const int E0    = tile * 16;
  const int col15 = lane & 15;
  const int half  = lane >> 4;

  const float* arow = Xe + (size_t)(E0 + col15) * EDGE_IN;

  v8f acc[8] = {};
#pragma unroll
  for (int kk = 0; kk < EDGE_IN / 32; ++kk) {
    v16bf af = load_a_frag_f32(arow, kk * 32, half);
#pragma unroll
    for (int t = 0; t < 8; ++t) {
      v16bf bf = load_b_frag_packed(sW, kk * 8 + t, lane);
      acc[t] = wmma_bf16(af, bf, acc[t]);
    }
  }

  // LayerNorm over HID (no GELU on the edge path), then exp.
  float s1[8], s2[8];
#pragma unroll
  for (int m = 0; m < 8; ++m) { s1[m] = 0.f; s2[m] = 0.f; }
#pragma unroll
  for (int t = 0; t < 8; ++t)
#pragma unroll
    for (int m = 0; m < 8; ++m) {
      float x = acc[t][m];
      s1[m] += x;
      s2[m] += x * x;
    }
#pragma unroll
  for (int m = 0; m < 8; ++m)
#pragma unroll
    for (int mask = 1; mask < 16; mask <<= 1) {
      s1[m] += __shfl_xor(s1[m], mask, 32);
      s2[m] += __shfl_xor(s2[m], mask, 32);
    }

  float mean[8], rstd[8];
#pragma unroll
  for (int m = 0; m < 8; ++m) {
    mean[m] = s1[m] * (1.0f / HID);
    float var = s2[m] * (1.0f / HID) - mean[m] * mean[m];
    rstd[m] = rsqrtf(var + LN_EPS);
  }

  float gg[8], bb[8];
#pragma unroll
  for (int t = 0; t < 8; ++t) {
    gg[t] = g[t * 16 + col15];
    bb[t] = b[t * 16 + col15];
  }

  // Gather hv[src] (hot in 192 MB L2), multiply, scatter-add into h[dst].
#pragma unroll
  for (int m = 0; m < 8; ++m) {
    const int e = E0 + m + half * 8;
    const int s = src[e];
    const int d = dst[e];
    const float* hvrow = hv + (size_t)s * HID;
    float*       hrow  = h + (size_t)d * HID;
#pragma unroll
    for (int t = 0; t < 8; ++t) {
      const int c  = t * 16 + col15;
      const float he = __expf((acc[t][m] - mean[m]) * rstd[m] * gg[t] + bb[t]);
      atomicAdd(&hrow[c], he * hvrow[c]);
    }
  }
}

// ---------------------------------------------------------------------------
// Kernel 3: out = LN(GELU(h @ W_out)), OUT=16 (single N-tile), K=128
// ---------------------------------------------------------------------------
__global__ void __launch_bounds__(256)
out_proj_kernel(const float* __restrict__ h, const float* __restrict__ W,
                const float* __restrict__ g, const float* __restrict__ b,
                float* __restrict__ out) {
  __shared__ alignas(32) __bf16 sW[HID * OUT_DIM];  // 4 KB
  stage_w_packed<HID, 1>(W, sW);
  __syncthreads();

  const int wave  = threadIdx.x >> 5;
  const int lane  = threadIdx.x & 31;
  const int tile  = blockIdx.x * 8 + wave;
  if (tile >= N_NODES / 16) return;
  const int R     = tile * 16;
  const int col15 = lane & 15;
  const int half  = lane >> 4;

  const float* arow = h + (size_t)(R + col15) * HID;

  v8f acc = {};
#pragma unroll
  for (int kk = 0; kk < HID / 32; ++kk) {
    v16bf af = load_a_frag_f32(arow, kk * 32, half);
    v16bf bf = load_b_frag_packed(sW, kk, lane);
    acc = wmma_bf16(af, bf, acc);
  }

  // GELU then LayerNorm over OUT=16 (one value per lane of the half-group).
#pragma unroll
  for (int m = 0; m < 8; ++m) {
    float x = gelu_exact(acc[m]);
    float s1 = x, s2 = x * x;
#pragma unroll
    for (int mask = 1; mask < 16; mask <<= 1) {
      s1 += __shfl_xor(s1, mask, 32);
      s2 += __shfl_xor(s2, mask, 32);
    }
    const float mean = s1 * (1.0f / OUT_DIM);
    const float var  = s2 * (1.0f / OUT_DIM) - mean * mean;
    const float rstd = rsqrtf(var + LN_EPS);
    const int r = R + m + half * 8;
    out[(size_t)r * OUT_DIM + col15] =
        (x - mean) * rstd * g[col15] + b[col15];
  }
}

// ---------------------------------------------------------------------------
extern "C" void kernel_launch(void* const* d_in, const int* in_sizes, int n_in,
                              void* d_out, int out_size, void* d_ws, size_t ws_size,
                              hipStream_t stream) {
  const float* node_feats = (const float*)d_in[0];
  const float* edge_feats = (const float*)d_in[1];
  const int*   src        = (const int*)d_in[2];
  const int*   dst        = (const int*)d_in[3];
  const float* W_node     = (const float*)d_in[4];
  const float* g_node     = (const float*)d_in[5];
  const float* b_node     = (const float*)d_in[6];
  const float* W_edge     = (const float*)d_in[7];
  const float* g_edge     = (const float*)d_in[8];
  const float* b_edge     = (const float*)d_in[9];
  const float* W_out      = (const float*)d_in[10];
  const float* g_out      = (const float*)d_in[11];
  const float* b_out      = (const float*)d_in[12];

  float* hv = (float*)d_ws;                       // 25.6 MB
  float* h  = hv + (size_t)N_NODES * HID;         // 25.6 MB

  hipMemsetAsync(h, 0, (size_t)N_NODES * HID * sizeof(float), stream);

  const int nodeTiles = N_NODES / 16;   // 3125 (exact)
  const int edgeTiles = N_EDGES / 16;   // 50000 (exact)

  node_proj_kernel<<<(nodeTiles + 7) / 8, 256, 0, stream>>>(
      node_feats, W_node, g_node, b_node, hv);
  edge_msg_kernel<<<edgeTiles / 8, 256, 0, stream>>>(
      edge_feats, src, dst, W_edge, g_edge, b_edge, hv, h);
  out_proj_kernel<<<(nodeTiles + 7) / 8, 256, 0, stream>>>(
      h, W_out, g_out, b_out, (float*)d_out);
}
